// SupervisedGraphsage_84997402788193
// MI455X (gfx1250) — compile-verified, matrix operand order checked by reading, not run on hardware
//
#include <hip/hip_runtime.h>

// GraphSAGE inference for MI455X (gfx1250, wave32, WMMA).
// Memory-bound on feature gathers (~290 MB @ 23.3 TB/s => ~12.5us floor);
// GEMMs run as f16 WMMA w/ f32 accumulate, weights staged in LDS per block.

#define KDIM   256   // all GEMM K dims (D and 2H) are 256
#define B_ROWS 1024
#define S1F    25
#define S2F    10
#define H_DIM  128
#define C_DIM  64

typedef _Float16 f16;
typedef __attribute__((ext_vector_type(16))) _Float16 v16h;
typedef __attribute__((ext_vector_type(8)))  _Float16 v8h;
typedef __attribute__((ext_vector_type(4)))  _Float16 v4h;
typedef __attribute__((ext_vector_type(8)))  float    v8f;

// Load one 16-half WMMA fragment for this lane. Per ISA 7.12.2 (16-bit A 16x32):
// halves 0..7 -> K=[kSh..kSh+8), halves 8..15 -> K=[kSh+16..kSh+24), kSh = 8*hi.
// Both runs are contiguous 16B -> b128 loads (global or ds, by address space).
__device__ __forceinline__ v16h load_frag(const f16* p) {
    v8h lo = *(const v8h*)(p);
    v8h hi = *(const v8h*)(p + 16);
    v16h r;
#pragma unroll
    for (int i = 0; i < 8; ++i) { r[i] = lo[i]; r[i + 8] = hi[i]; }
    return r;
}

__device__ __forceinline__ v8f wmma_f16f32(v16h a, v16h b, v8f c) {
    // (neg_a, A, neg_b, B, c_mod, C, reuse_a, reuse_b)
    return __builtin_amdgcn_wmma_f32_16x16x32_f16(false, a, false, b, (short)0, c,
                                                  false, false);
}

// W [KDIM x Ncols] f32 row-major -> Wt [Ncols x KDIM] f16 (column-major weights)
__global__ void transpose_w_f16(const float* __restrict__ W, f16* __restrict__ Wt,
                                int Ncols) {
    const int idx = blockIdx.x * blockDim.x + threadIdx.x;
    if (idx >= KDIM * Ncols) return;
    const int k = idx / Ncols;
    const int n = idx - k * Ncols;
    Wt[(size_t)n * KDIM + k] = (f16)W[idx];
}

// out[row, :] = f16( mean_{j<fan} features[ids[row*fan+j], :] ).  64 thr/row,
// 4 cols/thread via float4 (coalesced 1KB rows), v4h (8B) stores.
__global__ void gather_mean_to_f16(const int* __restrict__ ids,
                                   const float* __restrict__ feat,
                                   f16* __restrict__ out, int fan) {
    const int row = blockIdx.x;
    const int c   = threadIdx.x * 4;
    float ax = 0.f, ay = 0.f, az = 0.f, aw = 0.f;
    for (int j = 0; j < fan; ++j) {
        const int id = ids[row * fan + j];
        const float4 v = *(const float4*)(feat + (size_t)id * KDIM + c);
        ax += v.x; ay += v.y; az += v.z; aw += v.w;
    }
    const float s = 1.0f / (float)fan;
    v4h o;
    o[0] = (f16)(ax * s); o[1] = (f16)(ay * s);
    o[2] = (f16)(az * s); o[3] = (f16)(aw * s);
    *(v4h*)(out + (size_t)row * KDIM + c) = o;
}

// out[row, :] = f16( mean_{j<fan} in[row*fan+j, :] ), f16 in, f32 accumulate.
__global__ void mean_rows_f16(const f16* __restrict__ in, f16* __restrict__ out,
                              int fan) {
    const int row = blockIdx.x;
    const int c   = threadIdx.x * 4;
    float a0 = 0.f, a1 = 0.f, a2 = 0.f, a3 = 0.f;
    for (int j = 0; j < fan; ++j) {
        v4h v = *(const v4h*)(in + ((size_t)row * fan + j) * KDIM + c);
        a0 += (float)v[0]; a1 += (float)v[1];
        a2 += (float)v[2]; a3 += (float)v[3];
    }
    const float s = 1.0f / (float)fan;
    v4h o;
    o[0] = (f16)(a0 * s); o[1] = (f16)(a1 * s);
    o[2] = (f16)(a2 * s); o[3] = (f16)(a3 * s);
    *(v4h*)(out + (size_t)row * KDIM + c) = o;
}

// C[m, colOff + group*64 + t*16 + n] = act(A @ Wt^T + bias)
// A: [rows x 256] f16 row-major; Wt: [Nout x 256] f16 (pre-transposed weights).
// One block: 8 waves = 8 consecutive 16-row M-tiles x one 64-col N-group.
// The 64x256 f16 weight panel (32 KB) is staged in LDS once and shared by all
// 8 waves. Per K-step: load A (2x global b128) + all 4 B fragments (8x ds b128)
// into distinct registers, then issue 4 independent back-to-back WMMAs.
__global__ void __launch_bounds__(256)
gemm_wmma_k256(const f16* __restrict__ A, const f16* __restrict__ Wt,
               const float* __restrict__ bias,
               f16* __restrict__ outH, float* __restrict__ outF,
               int ldOut, int colOff, int nGroups, int doRelu) {
    __shared__ f16 Bp[64 * KDIM];   // 32 KB panel: rows n=0..63, 256 halves each

    const int tid    = threadIdx.x;
    const int lane   = tid & 31;
    const int wid    = tid >> 5;
    const int mBlock = blockIdx.x / nGroups;
    const int group  = blockIdx.x - mBlock * nGroups;
    const int mTile  = mBlock * 8 + wid;

    // Stage panel: 2048 16-byte chunks, 8 per thread, coalesced.
    {
        const f16* src = Wt + (size_t)group * 64 * KDIM;
#pragma unroll
        for (int i = 0; i < 8; ++i) {
            const int chunk = i * 256 + tid;
            *(v8h*)(Bp + chunk * 8) = *(const v8h*)(src + chunk * 8);
        }
    }
    __syncthreads();

    const int nl  = lane & 15;   // A row-in-tile == B/C column-in-tile
    const int hi  = lane >> 4;
    const int kSh = hi * 8;

    const f16* aRow = A + (size_t)(mTile * 16 + nl) * KDIM;
    const f16* bBase = Bp + (size_t)nl * KDIM;

    v8f acc[4];
#pragma unroll
    for (int t = 0; t < 4; ++t) acc[t] = (v8f){0.f,0.f,0.f,0.f,0.f,0.f,0.f,0.f};

#pragma unroll
    for (int kt = 0; kt < KDIM / 32; ++kt) {
        const int k0 = kt * 32 + kSh;
        const v16h a = load_frag(aRow + k0);      // reused across 4 B tiles
        v16h b[4];
#pragma unroll
        for (int t = 0; t < 4; ++t)               // one 8-wide ds_load clause
            b[t] = load_frag(bBase + t * 16 * KDIM + k0);
#pragma unroll
        for (int t = 0; t < 4; ++t)               // 4 independent WMMAs
            acc[t] = wmma_f16f32(a, b[t], acc[t]);
    }

    // C/D layout: lane n = lane&15; VGPR v -> M = v + 8*hi
#pragma unroll
    for (int t = 0; t < 4; ++t) {
        const int col = group * 64 + t * 16 + nl;
        const float bv = bias[col];
#pragma unroll
        for (int v = 0; v < 8; ++v) {
            const int mr = mTile * 16 + v + hi * 8;
            float val = acc[t][v] + bv;
            if (doRelu) val = fmaxf(val, 0.0f);
            if (outF) outF[(size_t)mr * ldOut + colOff + col] = val;
            else      outH[(size_t)mr * ldOut + colOff + col] = (f16)val;
        }
    }
}

extern "C" void kernel_launch(void* const* d_in, const int* in_sizes, int n_in,
                              void* d_out, int out_size, void* d_ws, size_t ws_size,
                              hipStream_t stream) {
    (void)in_sizes; (void)n_in; (void)out_size; (void)ws_size;

    const int*   ids0 = (const int*)d_in[0];
    const int*   ids1 = (const int*)d_in[1];
    const int*   ids2 = (const int*)d_in[2];
    const float* feat = (const float*)d_in[3];
    const float* Wx1  = (const float*)d_in[4];
    const float* bx1  = (const float*)d_in[5];
    const float* Wn1  = (const float*)d_in[6];
    const float* bn1  = (const float*)d_in[7];
    const float* Wx2  = (const float*)d_in[8];
    const float* bx2  = (const float*)d_in[9];
    const float* Wn2  = (const float*)d_in[10];
    const float* bn2  = (const float*)d_in[11];
    const float* Wfc  = (const float*)d_in[12];
    const float* bfc  = (const float*)d_in[13];
    float* out = (float*)d_out;

    // -------- workspace layout (~41 MB total, 256B aligned blocks) --------
    char* ws = (char*)d_ws;
    size_t off = 0;
    auto alloc = [&](size_t bytes) {
        char* p = ws + off;
        off += (bytes + 255) & ~(size_t)255;
        return p;
    };
    f16* Wx1t = (f16*)alloc((size_t)H_DIM * KDIM * sizeof(f16));
    f16* Wn1t = (f16*)alloc((size_t)H_DIM * KDIM * sizeof(f16));
    f16* Wx2t = (f16*)alloc((size_t)H_DIM * KDIM * sizeof(f16));
    f16* Wn2t = (f16*)alloc((size_t)H_DIM * KDIM * sizeof(f16));
    f16* Wfct = (f16*)alloc((size_t)C_DIM * KDIM * sizeof(f16));
    f16* X0 = (f16*)alloc((size_t)B_ROWS * KDIM * sizeof(f16));
    f16* N0 = (f16*)alloc((size_t)B_ROWS * KDIM * sizeof(f16));
    f16* X1 = (f16*)alloc((size_t)B_ROWS * S1F * KDIM * sizeof(f16));
    f16* N1 = (f16*)alloc((size_t)B_ROWS * S1F * KDIM * sizeof(f16));
    f16* G1 = (f16*)alloc((size_t)B_ROWS * S1F * KDIM * sizeof(f16));
    f16* G0 = (f16*)alloc((size_t)B_ROWS * KDIM * sizeof(f16));
    f16* M1 = (f16*)alloc((size_t)B_ROWS * KDIM * sizeof(f16));
    f16* F0 = (f16*)alloc((size_t)B_ROWS * KDIM * sizeof(f16));

    // -------- 1) weight transposes to f16 column-major --------
    transpose_w_f16<<<(KDIM * H_DIM + 255) / 256, 256, 0, stream>>>(Wx1, Wx1t, H_DIM);
    transpose_w_f16<<<(KDIM * H_DIM + 255) / 256, 256, 0, stream>>>(Wn1, Wn1t, H_DIM);
    transpose_w_f16<<<(KDIM * H_DIM + 255) / 256, 256, 0, stream>>>(Wx2, Wx2t, H_DIM);
    transpose_w_f16<<<(KDIM * H_DIM + 255) / 256, 256, 0, stream>>>(Wn2, Wn2t, H_DIM);
    transpose_w_f16<<<(KDIM * C_DIM + 255) / 256, 256, 0, stream>>>(Wfc, Wfct, C_DIM);

    // -------- 2) gathers (+ neighbor means) into f16 --------
    const int R1 = B_ROWS * S1F;  // 25600
    gather_mean_to_f16<<<B_ROWS, 64, 0, stream>>>(ids0, feat, X0, 1);
    gather_mean_to_f16<<<R1,     64, 0, stream>>>(ids1, feat, X1, 1);
    gather_mean_to_f16<<<B_ROWS, 64, 0, stream>>>(ids1, feat, N0, S1F);
    gather_mean_to_f16<<<R1,     64, 0, stream>>>(ids2, feat, N1, S2F);

    // blocks = (rows/128) * nGroups  (8 M-tiles per block, 64 cols per group)
    // -------- 3) layer-1 GEMMs, relu, concat halves --------
    gemm_wmma_k256<<<(R1 / 128) * 2, 256, 0, stream>>>(
        X1, Wx1t, bx1, G1, nullptr, 2 * H_DIM, 0,     2, 1);
    gemm_wmma_k256<<<(R1 / 128) * 2, 256, 0, stream>>>(
        N1, Wn1t, bn1, G1, nullptr, 2 * H_DIM, H_DIM, 2, 1);
    gemm_wmma_k256<<<(B_ROWS / 128) * 2, 256, 0, stream>>>(
        X0, Wx1t, bx1, G0, nullptr, 2 * H_DIM, 0,     2, 1);
    gemm_wmma_k256<<<(B_ROWS / 128) * 2, 256, 0, stream>>>(
        N0, Wn1t, bn1, G0, nullptr, 2 * H_DIM, H_DIM, 2, 1);

    // -------- 4) mean of g1 over the S1 fanout --------
    mean_rows_f16<<<B_ROWS, 64, 0, stream>>>(G1, M1, S1F);

    // -------- 5) layer-2 GEMMs (identity activation) --------
    gemm_wmma_k256<<<(B_ROWS / 128) * 2, 256, 0, stream>>>(
        G0, Wx2t, bx2, F0, nullptr, 2 * H_DIM, 0,     2, 0);
    gemm_wmma_k256<<<(B_ROWS / 128) * 2, 256, 0, stream>>>(
        M1, Wn2t, bn2, F0, nullptr, 2 * H_DIM, H_DIM, 2, 0);

    // -------- 6) classifier head -> f32 output [1024 x 64] --------
    gemm_wmma_k256<<<(B_ROWS / 128) * 1, 256, 0, stream>>>(
        F0, Wfct, bfc, nullptr, out, C_DIM, 0, 1, 0);
}
